// HyperTransformer_21861383537287
// MI455X (gfx1250) — matrix operand order repaired and checked by reading, not compiled, not run on hardware
//
#include <hip/hip_runtime.h>
#include <hip/hip_bf16.h>
#include <math.h>

// ---------------------------------------------------------------------------
// HyperTransformer forward for MI455X (gfx1250, wave32, WMMA).
// GEMMs: 128x64 block tile, 8 waves x (32x32 register tile) = 4 WMMAs/wave
// per K-step, LDS staged in fragment-swizzled f16 so fragment reads are
// ds_load_b128 pairs; double-buffered tiles, float4 global staging.
// Attention: flash-style, one wave per 16-query tile per head.
// Hypergraph scatter: coalesced float4 gathers + global_atomic_add_f32.
// ---------------------------------------------------------------------------

typedef __attribute__((ext_vector_type(16))) _Float16 v16h;
typedef __attribute__((ext_vector_type(8)))  float    v8f;

#define N_NODES 4096
#define N_EDGES 8192
#define NNZ     524288
#define IN_CH   128
#define HID     256
#define OUT_CH  64

// ---------------------------------------------------------------------------
// Tiled WMMA GEMM: C[M,N] = act(A[M,K] @ B[K,N] + bias)
// Requires M%128==0, N%64==0; K arbitrary (zero-padded in LDS).
// LDS layouts hold fragments directly:
//   As[buf][rowTile][lane][e]  == A(bm + rowTile*16 + (lane&15), k0 + k(e,g))
//     with g = lane>>4, k(e,g) = (e>=8 ? 16 : 0) + 8*g + (e&7)
//   Bs[buf][colTile][lane][e]  == B(k0 + k(e,g), bn + colTile*16 + (lane&15))
// so a fragment is one contiguous 32-byte v16h per lane (2x ds_load_b128).
// ---------------------------------------------------------------------------
template <bool RELU, bool HASBIAS>
__global__ __launch_bounds__(256) void gemm_wmma_kernel(
    const float* __restrict__ A, const float* __restrict__ B,
    const float* __restrict__ bias, float* __restrict__ C,
    int M, int N, int K) {
  const int BM = 128, BN = 64, BK = 32;
  __shared__ __attribute__((aligned(32))) _Float16 As[2][8][32][16];  // 16 KB
  __shared__ __attribute__((aligned(32))) _Float16 Bs[2][4][32][16];  // 8 KB

  const int bm = blockIdx.x * BM;
  const int bn = blockIdx.y * BN;
  const int tid = threadIdx.x;
  const int wave = tid >> 5, lane = tid & 31;
  const int wr = wave >> 1, wc = wave & 1;  // wave grid 4x2 of 32x32 tiles
  const int g = lane >> 4, nl = lane & 15;

  // Staging assignments (per thread): A row + 16 consecutive k's; B k + 8 n's.
  const int ar = tid >> 1, ac0 = (tid & 1) * 16;
  const int bk = tid >> 3, bc0 = (tid & 7) * 8;

  auto stage = [&](int buf, int k0) {
    if (k0 + BK <= K) {
      // Fast path: vector loads (global_load_b128), packed f16 LDS stores.
#pragma unroll
      for (int q = 0; q < 4; ++q) {
        int c = ac0 + q * 4;
        const float4 v = *(const float4*)&A[(size_t)(bm + ar) * K + k0 + c];
        _Float16 h4[4] = {(_Float16)v.x, (_Float16)v.y, (_Float16)v.z,
                          (_Float16)v.w};
        int lg = (c >> 3) & 1;
        int e0 = ((c & 16) ? 8 : 0) + (c & 7);
        *(uint2*)&As[buf][ar >> 4][(ar & 15) + 16 * lg][e0] =
            *(const uint2*)h4;
      }
      const float4 v0 = *(const float4*)&B[(size_t)(k0 + bk) * N + bn + bc0];
      const float4 v1 =
          *(const float4*)&B[(size_t)(k0 + bk) * N + bn + bc0 + 4];
      const float bv[8] = {v0.x, v0.y, v0.z, v0.w, v1.x, v1.y, v1.z, v1.w};
#pragma unroll
      for (int q = 0; q < 8; ++q) {
        int c = bc0 + q;
        Bs[buf][c >> 4][(c & 15) + 16 * ((bk >> 3) & 1)][((bk & 16) ? 8 : 0) + (bk & 7)] = (_Float16)bv[q];
      }
    } else {
      // Tail path (K not multiple of 32): guarded scalar loads, zero pad.
#pragma unroll
      for (int q = 0; q < 16; ++q) {
        int c = ac0 + q;
        int gk = k0 + c;
        float v = (gk < K) ? A[(size_t)(bm + ar) * K + gk] : 0.0f;
        int lg = (c >> 3) & 1;
        int e = ((c & 16) ? 8 : 0) + (c & 7);
        As[buf][ar >> 4][(ar & 15) + 16 * lg][e] = (_Float16)v;
      }
#pragma unroll
      for (int q = 0; q < 8; ++q) {
        int c = bc0 + q;
        int gk = k0 + bk;
        float v = (gk < K) ? B[(size_t)gk * N + bn + c] : 0.0f;
        Bs[buf][c >> 4][(c & 15) + 16 * ((bk >> 3) & 1)][((bk & 16) ? 8 : 0) + (bk & 7)] = (_Float16)v;
      }
    }
  };

  v8f acc[2][2];
  {
    v8f z = {};
#pragma unroll
    for (int i = 0; i < 2; ++i)
#pragma unroll
      for (int j = 0; j < 2; ++j) acc[i][j] = z;
  }

  stage(0, 0);
  __syncthreads();
  const int niter = (K + BK - 1) / BK;
  for (int it = 0; it < niter; ++it) {
    const int buf = it & 1;
    const int k0 = it * BK;
    if (k0 + BK < K) {
      if (k0 + 2 * BK < K)  // pull the tile after next (global_prefetch_b8)
        __builtin_prefetch(&A[(size_t)(bm + ar) * K + k0 + 2 * BK + ac0], 0, 0);
      stage(buf ^ 1, k0 + BK);  // writes other buffer; overlaps compute
    }
    v16h a0 = *(const v16h*)&As[buf][2 * wr + 0][lane][0];
    v16h a1 = *(const v16h*)&As[buf][2 * wr + 1][lane][0];
    v16h b0 = *(const v16h*)&Bs[buf][2 * wc + 0][lane][0];
    v16h b1 = *(const v16h*)&Bs[buf][2 * wc + 1][lane][0];
    acc[0][0] = __builtin_amdgcn_wmma_f32_16x16x32_f16(false, a0, false, b0,
                                                       (short)0, acc[0][0],
                                                       false, false);
    acc[0][1] = __builtin_amdgcn_wmma_f32_16x16x32_f16(false, a0, false, b1,
                                                       (short)0, acc[0][1],
                                                       false, false);
    acc[1][0] = __builtin_amdgcn_wmma_f32_16x16x32_f16(false, a1, false, b0,
                                                       (short)0, acc[1][0],
                                                       false, false);
    acc[1][1] = __builtin_amdgcn_wmma_f32_16x16x32_f16(false, a1, false, b1,
                                                       (short)0, acc[1][1],
                                                       false, false);
    __syncthreads();
  }

  // Epilogue: C-layout (row = i + 8*g, col = nl), coalesced across lanes.
#pragma unroll
  for (int ti = 0; ti < 2; ++ti)
#pragma unroll
    for (int tj = 0; tj < 2; ++tj) {
      const int gn = bn + (2 * wc + tj) * 16 + nl;
      const float bv = HASBIAS ? bias[gn] : 0.0f;
#pragma unroll
      for (int i = 0; i < 8; ++i) {
        const int gm = bm + (2 * wr + ti) * 16 + i + 8 * g;
        float v = acc[ti][tj][i] + bv;
        if (RELU) v = fmaxf(v, 0.0f);
        C[(size_t)gm * N + gn] = v;
      }
    }
}

// ---------------------------------------------------------------------------
// Row LayerNorm with optional residual: Y = LN(X + R) * gam + bet.
// ---------------------------------------------------------------------------
__global__ __launch_bounds__(256) void ln_kernel(
    const float* __restrict__ X, const float* __restrict__ R,
    const float* __restrict__ gam, const float* __restrict__ bet,
    float* __restrict__ Y, int C) {
  const int row = blockIdx.x, tid = threadIdx.x;
  const size_t base = (size_t)row * C;
  __shared__ float red[256];

  float s = 0.0f;
  for (int c = tid; c < C; c += 256) {
    float v = X[base + c] + (R ? R[base + c] : 0.0f);
    Y[base + c] = v;
    s += v;
  }
  red[tid] = s;
  __syncthreads();
  for (int o = 128; o > 0; o >>= 1) {
    if (tid < o) red[tid] += red[tid + o];
    __syncthreads();
  }
  float mu = red[0] / (float)C;
  __syncthreads();

  float vs = 0.0f;
  for (int c = tid; c < C; c += 256) {
    float d = Y[base + c] - mu;
    vs += d * d;
  }
  red[tid] = vs;
  __syncthreads();
  for (int o = 128; o > 0; o >>= 1) {
    if (tid < o) red[tid] += red[tid + o];
    __syncthreads();
  }
  float rs = rsqrtf(red[0] / (float)C + 1e-5f);

  for (int c = tid; c < C; c += 256)
    Y[base + c] = (Y[base + c] - mu) * rs * gam[c] + bet[c];
}

// ---------------------------------------------------------------------------
// Flash attention: 2 heads, dh=128, seq N. One wave per (16-query tile, head).
// QK^T and PV via v_wmma_f32_16x16x32_f16; LDS only for P layout transpose.
// ---------------------------------------------------------------------------
__global__ __launch_bounds__(32) void flash_attn_kernel(
    const float* __restrict__ qkv, float* __restrict__ out, int N) {
  const int q0 = blockIdx.x * 16;
  const int hd = blockIdx.y * 128;
  const int lane = threadIdx.x;
  const int g = lane >> 4;
  const float scale = 0.08838834764831845f;  // 1/sqrt(128)

  __shared__ _Float16 Plds[16 * 32];

  v16h qf[4];
  {
    int m = lane & 15;
#pragma unroll
    for (int sl = 0; sl < 4; ++sl)
#pragma unroll
      for (int e = 0; e < 16; ++e) {
        int v = e >> 1;
        int k = 16 * g + e;
        qf[sl][e] =
            (_Float16)(qkv[(size_t)(q0 + m) * 768 + hd + sl * 32 + k] * scale);
      }
  }

  v8f o[8];
#pragma unroll
  for (int c = 0; c < 8; ++c)
#pragma unroll
    for (int i = 0; i < 8; ++i) o[c][i] = 0.0f;
  float mrow[8], lrow[8];
#pragma unroll
  for (int i = 0; i < 8; ++i) { mrow[i] = -1e30f; lrow[i] = 0.0f; }

  for (int kt = 0; kt < N; kt += 32) {
    v8f S0 = {}, S1 = {};
#pragma unroll
    for (int sl = 0; sl < 4; ++sl) {
      v16h b0, b1;
      int n = lane & 15, kb = g * 16;
#pragma unroll
      for (int e = 0; e < 16; ++e) {
        int d = sl * 32 + kb + e;
        b0[e] = (_Float16)qkv[(size_t)(kt + n) * 768 + 256 + hd + d];
        b1[e] = (_Float16)qkv[(size_t)(kt + 16 + n) * 768 + 256 + hd + d];
      }
      S0 = __builtin_amdgcn_wmma_f32_16x16x32_f16(false, qf[sl], false, b0,
                                                  (short)0, S0, false, false);
      S1 = __builtin_amdgcn_wmma_f32_16x16x32_f16(false, qf[sl], false, b1,
                                                  (short)0, S1, false, false);
    }

    float corr[8];
#pragma unroll
    for (int i = 0; i < 8; ++i) {
      float mx = fmaxf(S0[i], S1[i]);
#pragma unroll
      for (int off = 1; off < 16; off <<= 1)
        mx = fmaxf(mx, __shfl_xor(mx, off, 32));
      float mn = fmaxf(mrow[i], mx);
      corr[i] = __expf(mrow[i] - mn);
      float p0 = __expf(S0[i] - mn), p1 = __expf(S1[i] - mn);
      float ps = p0 + p1;
#pragma unroll
      for (int off = 1; off < 16; off <<= 1) ps += __shfl_xor(ps, off, 32);
      lrow[i] = lrow[i] * corr[i] + ps;
      mrow[i] = mn;
      Plds[(i + 8 * g) * 32 + (lane & 15)] = (_Float16)p0;
      Plds[(i + 8 * g) * 32 + 16 + (lane & 15)] = (_Float16)p1;
    }
#pragma unroll
    for (int c = 0; c < 8; ++c)
#pragma unroll
      for (int i = 0; i < 8; ++i) o[c][i] *= corr[i];
    __syncthreads();

    v16h pf;
    {
      int m = lane & 15;
#pragma unroll
      for (int e = 0; e < 16; ++e) {
        int v = e >> 1;
        int k = 16 * g + e;
        pf[e] = Plds[m * 32 + k];
      }
    }
#pragma unroll
    for (int c = 0; c < 8; ++c) {
      v16h vf;
      int n = lane & 15, kb = g * 16;
#pragma unroll
      for (int e = 0; e < 16; ++e)
        vf[e] =
            (_Float16)qkv[(size_t)(kt + kb + e) * 768 + 512 + hd + c * 16 + n];
      o[c] = __builtin_amdgcn_wmma_f32_16x16x32_f16(false, pf, false, vf,
                                                    (short)0, o[c], false,
                                                    false);
    }
    __syncthreads();
  }

#pragma unroll
  for (int c = 0; c < 8; ++c)
#pragma unroll
    for (int i = 0; i < 8; ++i)
      out[(size_t)(q0 + i + 8 * g) * 256 + hd + c * 16 + (lane & 15)] =
          o[c][i] / lrow[i];
}

// ---------------------------------------------------------------------------
// Small helper kernels
// ---------------------------------------------------------------------------
__global__ void fill_kernel(float* __restrict__ p, float v, int n) {
  int i = blockIdx.x * blockDim.x + threadIdx.x;
  if (i < n) p[i] = v;
}

__global__ __launch_bounds__(256) void rowdot_kernel(
    const float* __restrict__ X, const float* __restrict__ w,
    float* __restrict__ out, int C) {
  const int row = blockIdx.x, tid = threadIdx.x;
  __shared__ float red[256];
  float s = 0.0f;
  for (int c = tid; c < C; c += 256) s += X[(size_t)row * C + c] * w[c];
  red[tid] = s;
  __syncthreads();
  for (int o = 128; o > 0; o >>= 1) {
    if (tid < o) red[tid] += red[tid + o];
    __syncthreads();
  }
  if (tid == 0) out[row] = red[0];
}

__device__ inline void atomicMaxF(float* addr, float val) {
  unsigned int* ua = (unsigned int*)addr;
  unsigned int old = *ua;
  while (__uint_as_float(old) < val) {
    unsigned int assumed = old;
    old = atomicCAS(ua, assumed, __float_as_uint(val));
    if (old == assumed) break;
  }
}

__global__ void alpha_max_kernel(const int* __restrict__ row,
                                 const int* __restrict__ col,
                                 const float* __restrict__ ax,
                                 const float* __restrict__ ae,
                                 float* __restrict__ alpha,
                                 float* __restrict__ segmax, int nnz) {
  int i = blockIdx.x * blockDim.x + threadIdx.x;
  if (i >= nnz) return;
  float a = ax[row[i]] + ae[col[i]];
  a = (a > 0.0f) ? a : 0.2f * a;  // leaky_relu(0.2)
  alpha[i] = a;
  atomicMaxF(&segmax[col[i]], a);
}

__global__ void alpha_exp_kernel(const int* __restrict__ row,
                                 const int* __restrict__ col,
                                 float* __restrict__ alpha,
                                 const float* __restrict__ segmax,
                                 float* __restrict__ segsum,
                                 float* __restrict__ Dcnt,
                                 float* __restrict__ Bcnt, int nnz) {
  int i = blockIdx.x * blockDim.x + threadIdx.x;
  if (i >= nnz) return;
  int c = col[i];
  float a = __expf(alpha[i] - segmax[c]);
  alpha[i] = a;
  atomicAdd(&segsum[c], a);
  atomicAdd(&Dcnt[row[i]], 1.0f);
  atomicAdd(&Bcnt[c], 1.0f);
}

__global__ void alpha_norm_kernel(const int* __restrict__ col,
                                  float* __restrict__ alpha,
                                  const float* __restrict__ segsum, int nnz) {
  int i = blockIdx.x * blockDim.x + threadIdx.x;
  if (i >= nnz) return;
  alpha[i] = alpha[i] / segsum[col[i]];
}

// dst[idxd[i]] += alpha[i] * src[idxs[i]]  (256 channels, 4/thread, float4)
__global__ __launch_bounds__(256) void scatter_kernel(
    const int* __restrict__ idxs, const int* __restrict__ idxd,
    const float* __restrict__ alpha, const float* __restrict__ src,
    float* __restrict__ dst, int nnz) {
  size_t t = (size_t)blockIdx.x * blockDim.x + threadIdx.x;
  if (t >= (size_t)nnz * 64) return;
  int i = (int)(t >> 6);
  int c4 = (int)(t & 63) * 4;
  float a = alpha[i];
  const float4 v = *(const float4*)&src[(size_t)idxs[i] * 256 + c4];
  float* d = &dst[(size_t)idxd[i] * 256 + c4];
  atomicAdd(d + 0, a * v.x);
  atomicAdd(d + 1, a * v.y);
  atomicAdd(d + 2, a * v.z);
  atomicAdd(d + 3, a * v.w);
}

// X[r][c] = X[r][c] * (cnt[r]>0 ? 1/cnt[r] : 0) (+ bias[c] if given)
__global__ void scale_rows_kernel(float* __restrict__ X,
                                  const float* __restrict__ cnt,
                                  const float* __restrict__ bias, int rows,
                                  int C) {
  size_t idx = (size_t)blockIdx.x * blockDim.x + threadIdx.x;
  if (idx >= (size_t)rows * C) return;
  int r = (int)(idx / C), c = (int)(idx % C);
  float b = cnt[r];
  float s = (b > 0.0f) ? 1.0f / b : 0.0f;
  float v = X[idx] * s;
  if (bias) v += bias[c];
  X[idx] = v;
}

__global__ __launch_bounds__(256) void pool_kernel(const float* __restrict__ H,
                                                   float* __restrict__ out,
                                                   int Nn, int C) {
  int c = blockIdx.x * blockDim.x + threadIdx.x;
  if (c >= C) return;
  float s = 0.0f, mx = -1e30f;
  for (int r = 0; r < Nn; ++r) {
    float v = H[(size_t)r * C + c];
    s += v;
    mx = fmaxf(mx, v);
  }
  out[c] = s / (float)Nn + 0.5f * mx;
}

__global__ void gemv_kernel(const float* __restrict__ x,
                            const float* __restrict__ W,
                            const float* __restrict__ bias,
                            float* __restrict__ y, int K, int N, int relu) {
  int j = blockIdx.x * blockDim.x + threadIdx.x;
  if (j >= N) return;
  float s = bias[j];
  for (int k = 0; k < K; ++k) s += x[k] * W[(size_t)k * N + j];
  if (relu) s = fmaxf(s, 0.0f);
  y[j] = s;
}

// ---------------------------------------------------------------------------
// Host launcher
// ---------------------------------------------------------------------------
extern "C" void kernel_launch(void* const* d_in, const int* in_sizes, int n_in,
                              void* d_out, int out_size, void* d_ws,
                              size_t ws_size, hipStream_t stream) {
  (void)in_sizes; (void)n_in; (void)out_size; (void)ws_size;

  const float* x    = (const float*)d_in[0];
  const int*   hrow = (const int*)d_in[1];  // [2, NNZ] stacked
  const int*   hcol = hrow + NNZ;
  const float* attr = (const float*)d_in[2];
  const float* emb_w = (const float*)d_in[3];
  const float* emb_b = (const float*)d_in[4];
  const float* emb_g = (const float*)d_in[5];
  const float* emb_be= (const float*)d_in[6];
  const float* edg_w = (const float*)d_in[7];
  const float* edg_b = (const float*)d_in[8];
  const float* edg_g = (const float*)d_in[9];
  const float* edg_be= (const float*)d_in[10];
  const float* hg_w  = (const float*)d_in[11];
  const float* hg_att= (const float*)d_in[12];
  const float* hg_b  = (const float*)d_in[13];
  const float* ln1_g = (const float*)d_in[14];
  const float* ln1_b = (const float*)d_in[15];
  const float* qkv_w = (const float*)d_in[16];
  const float* qkv_b = (const float*)d_in[17];
  const float* aow   = (const float*)d_in[18];
  const float* aob   = (const float*)d_in[19];
  const float* tln1g = (const float*)d_in[20];
  const float* tln1b = (const float*)d_in[21];
  const float* f1w   = (const float*)d_in[22];
  const float* f1b   = (const float*)d_in[23];
  const float* f2w   = (const float*)d_in[24];
  const float* f2b   = (const float*)d_in[25];
  const float* tln2g = (const float*)d_in[26];
  const float* tln2b = (const float*)d_in[27];
  const float* ln2_g = (const float*)d_in[28];
  const float* ln2_b = (const float*)d_in[29];
  const float* prj_w = (const float*)d_in[30];
  const float* prj_b = (const float*)d_in[31];
  const float* prj_g = (const float*)d_in[32];
  const float* prj_be= (const float*)d_in[33];
  const float* fc_w  = (const float*)d_in[34];
  const float* fc_b  = (const float*)d_in[35];

  // Workspace layout (floats), ~61 MB with explicit reuse.
  float* ws = (float*)d_ws;
  const size_t SZ_N = (size_t)N_NODES * HID;
  const size_t SZ_E = (size_t)N_EDGES * HID;
  float* h     = ws;
  float* e     = h + SZ_N;
  float* el    = e + SZ_E;
  float* mid   = e;                  // [4096,1024] overlays e..el (dead)
  float* xl    = el + SZ_E;
  float* aot   = xl;                 // reuse (xl dead after scatter)
  float* out_e = xl + SZ_N;
  float* ao    = out_e;              // reuse front half
  float* ffo   = out_e + SZ_N;       // reuse back half
  float* out_n = out_e + SZ_E;
  float* xt    = out_n;              // reuse
  float* h2    = out_n + SZ_N;
  float* qkv   = h2 + SZ_N;
  float* t     = qkv + (size_t)N_NODES * 3 * HID;
  float* alpha = t + SZ_N;
  float* axv   = alpha + NNZ;
  float* aev   = axv + N_NODES;
  float* smax  = aev + N_EDGES;
  float* ssum  = smax + N_EDGES;
  float* Dcnt  = ssum + N_EDGES;
  float* Bcnt  = Dcnt + N_NODES;
  float* pooled= Bcnt + N_EDGES;
  float* o1    = pooled + HID;
  float* h3    = h;                  // reuse (h dead after h2)

  const dim3 B256(256), B32(32);

  // ---- node & edge embeddings --------------------------------------------
  gemm_wmma_kernel<true, true><<<dim3(N_NODES / 128, HID / 64), B256, 0,
                                 stream>>>(x, emb_w, emb_b, h, N_NODES, HID,
                                           IN_CH);
  ln_kernel<<<N_NODES, B256, 0, stream>>>(h, nullptr, emb_g, emb_be, h, HID);

  gemm_wmma_kernel<true, true><<<dim3(N_EDGES / 128, HID / 64), B256, 0,
                                 stream>>>(attr, edg_w, edg_b, e, N_EDGES, HID,
                                           6);
  ln_kernel<<<N_EDGES, B256, 0, stream>>>(e, nullptr, edg_g, edg_be, e, HID);

  // ---- hypergraph conv ----------------------------------------------------
  gemm_wmma_kernel<false, false><<<dim3(N_NODES / 128, HID / 64), B256, 0,
                                   stream>>>(h, hg_w, nullptr, xl, N_NODES,
                                             HID, HID);
  gemm_wmma_kernel<false, false><<<dim3(N_EDGES / 128, HID / 64), B256, 0,
                                   stream>>>(e, hg_w, nullptr, el, N_EDGES,
                                             HID, HID);
  rowdot_kernel<<<N_NODES, B256, 0, stream>>>(xl, hg_att, axv, HID);
  rowdot_kernel<<<N_EDGES, B256, 0, stream>>>(el, hg_att + HID, aev, HID);

  fill_kernel<<<(N_EDGES + 255) / 256, B256, 0, stream>>>(smax, -1e30f,
                                                          N_EDGES);
  fill_kernel<<<(N_EDGES + 255) / 256, B256, 0, stream>>>(ssum, 0.0f, N_EDGES);
  fill_kernel<<<(N_NODES + 255) / 256, B256, 0, stream>>>(Dcnt, 0.0f, N_NODES);
  fill_kernel<<<(N_EDGES + 255) / 256, B256, 0, stream>>>(Bcnt, 0.0f, N_EDGES);
  fill_kernel<<<(int)(SZ_E / 256), B256, 0, stream>>>(out_e, 0.0f, (int)SZ_E);
  fill_kernel<<<(int)(SZ_N / 256), B256, 0, stream>>>(out_n, 0.0f, (int)SZ_N);

  alpha_max_kernel<<<NNZ / 256, B256, 0, stream>>>(hrow, hcol, axv, aev, alpha,
                                                   smax, NNZ);
  alpha_exp_kernel<<<NNZ / 256, B256, 0, stream>>>(hrow, hcol, alpha, smax,
                                                   ssum, Dcnt, Bcnt, NNZ);
  alpha_norm_kernel<<<NNZ / 256, B256, 0, stream>>>(hcol, alpha, ssum, NNZ);

  scatter_kernel<<<NNZ / 4, B256, 0, stream>>>(hrow, hcol, alpha, xl, out_e,
                                               NNZ);
  scale_rows_kernel<<<(int)(SZ_E / 256), B256, 0, stream>>>(out_e, Bcnt,
                                                            nullptr, N_EDGES,
                                                            HID);
  scatter_kernel<<<NNZ / 4, B256, 0, stream>>>(hcol, hrow, alpha, out_e, out_n,
                                               NNZ);
  scale_rows_kernel<<<(int)(SZ_N / 256), B256, 0, stream>>>(out_n, Dcnt, hg_b,
                                                            N_NODES, HID);
  ln_kernel<<<N_NODES, B256, 0, stream>>>(h, out_n, ln1_g, ln1_b, h2, HID);

  // ---- transformer layer --------------------------------------------------
  gemm_wmma_kernel<false, true><<<dim3(N_NODES / 128, (3 * HID) / 64), B256, 0,
                                  stream>>>(h2, qkv_w, qkv_b, qkv, N_NODES,
                                            3 * HID, HID);
  flash_attn_kernel<<<dim3(N_NODES / 16, 2), B32, 0, stream>>>(qkv, aot,
                                                               N_NODES);
  gemm_wmma_kernel<false, true><<<dim3(N_NODES / 128, HID / 64), B256, 0,
                                  stream>>>(aot, aow, aob, ao, N_NODES, HID,
                                            HID);
  ln_kernel<<<N_NODES, B256, 0, stream>>>(h2, ao, tln1g, tln1b, t, HID);

  gemm_wmma_kernel<true, true><<<dim3(N_NODES / 128, (4 * HID) / 64), B256, 0,
                                 stream>>>(t, f1w, f1b, mid, N_NODES, 4 * HID,
                                           HID);
  gemm_wmma_kernel<false, true><<<dim3(N_NODES / 128, HID / 64), B256, 0,
                                  stream>>>(mid, f2w, f2b, ffo, N_NODES, HID,
                                            4 * HID);
  ln_kernel<<<N_NODES, B256, 0, stream>>>(t, ffo, tln2g, tln2b, xt, HID);
  ln_kernel<<<N_NODES, B256, 0, stream>>>(h2, xt, ln2_g, ln2_b, h3, HID);

  // ---- pooled head --------------------------------------------------------
  pool_kernel<<<1, B256, 0, stream>>>(h3, pooled, N_NODES, HID);
  gemv_kernel<<<1, B256, 0, stream>>>(pooled, prj_w, prj_b, o1, HID, HID, 1);
  ln_kernel<<<1, B256, 0, stream>>>(o1, nullptr, prj_g, prj_be, o1, HID);
  gemv_kernel<<<1, dim3(64), 0, stream>>>(o1, fc_w, fc_b, (float*)d_out, HID,
                                          OUT_CH, 0);
}
